// VQEmbedding_86672440033839
// MI455X (gfx1250) — matrix operand order repaired and checked by reading, not compiled, not run on hardware
//
#include <hip/hip_runtime.h>

// ---------------------------------------------------------------------------
// VQ nearest-codebook quantization for MI455X (gfx1250, wave32, WMMA).
//   z_e_x : [524288, 64] fp32      emb : [512, 64] fp32
//   out   : [2, 524288, 64] fp32 (both halves = gathered codes)
// Distance matmul runs on v_wmma_f32_16x16x32_bf16; outputs gathered in fp32.
// Memory-bound target: ~400 MB HBM traffic -> ~17 us at 23.3 TB/s.
// NOTE: every local-array subscript must be compile-time constant, else the
// array is demoted to scratch (observed in round 2: predicated scratch stores).
// ---------------------------------------------------------------------------

typedef __attribute__((ext_vector_type(16))) __bf16          v16bf;
typedef __attribute__((ext_vector_type(16))) unsigned short  v16u;
typedef __attribute__((ext_vector_type(4)))  unsigned short  v4u;
typedef __attribute__((ext_vector_type(8)))  float           v8f;

#define KCODES      512
#define DIM         64
#define EMB_STRIDE  72              // ushorts per LDS row (64 + 8 pad) -> bank-conflict free
#define BLOCK_THREADS 256
#define ROWS_PER_BLOCK 128          // 8 waves * 16 rows

__device__ __forceinline__ unsigned short f2bf(float f) {
    // round-to-nearest-even fp32 -> bf16 (bit manipulation: no fptrunc needed)
    unsigned int u = __float_as_uint(f);
    unsigned int r = u + 0x7FFFu + ((u >> 16) & 1u);
    return (unsigned short)(r >> 16);
}

extern "C" __global__ __launch_bounds__(BLOCK_THREADS)
void vq_argmin_gather_kernel(const float* __restrict__ z,
                             const float* __restrict__ emb,
                             float* __restrict__ out,
                             int nRows) {
    extern __shared__ char smem[];
    unsigned short* embLds = (unsigned short*)smem;                       // 512*72 u16
    float*          enorm  = (float*)(smem + KCODES * EMB_STRIDE * 2);    // 512 f32

    const int tid = threadIdx.x;

    // ---- stage codebook: fp32 -> bf16 LDS (vectorized: b128 loads, b64 DS stores)
    for (int i4 = tid; i4 < (KCODES * DIM) / 4; i4 += BLOCK_THREADS) {
        int i = i4 * 4;
        int n = i >> 6, k = i & 63;                  // float4 never crosses a row (64%4==0)
        float4 q = *(const float4*)(emb + i);
        v4u p = { f2bf(q.x), f2bf(q.y), f2bf(q.z), f2bf(q.w) };
        *(v4u*)(embLds + n * EMB_STRIDE + k) = p;    // 8B-aligned ds_store_b64
    }
    for (int n = tid; n < KCODES; n += BLOCK_THREADS) {
        float s = 0.f;
        #pragma unroll
        for (int k4 = 0; k4 < DIM / 4; ++k4) {
            float4 q = *(const float4*)(emb + n * DIM + k4 * 4);
            s += q.x * q.x + q.y * q.y + q.z * q.z + q.w * q.w;
        }
        enorm[n] = s;
    }
    __syncthreads();

    const int wave = tid >> 5;
    const int lane = tid & 31;
    const int li   = lane & 15;      // M (A) / N (B) position within tile
    const int h    = lane >> 4;      // half-wave selects K sub-range

    const long long row0 = (long long)blockIdx.x * ROWS_PER_BLOCK + wave * 16;
    const float* zrow = z + (row0 + li) * DIM;

    // ---- build A fragments (16-bit A 16x32 layout matching B: half h covers K in
    //      [16h,16h+16) per 32-wide chunk; within-lane order is contiguous).
    //      Each row byte read exactly once.
    float af[32];
    #pragma unroll
    for (int r = 0; r < 4; ++r) {                    // runs at K = (r>>1)*32 + 16h + (r&1)*8
        const float4* p = (const float4*)(zrow + (r >> 1) * 32 + 16 * h + (r & 1) * 8);
        float4 q0 = p[0], q1 = p[1];
        af[r*8+0]=q0.x; af[r*8+1]=q0.y; af[r*8+2]=q0.z; af[r*8+3]=q0.w;
        af[r*8+4]=q1.x; af[r*8+5]=q1.y; af[r*8+6]=q1.z; af[r*8+7]=q1.w;
    }
    v16u au0, au1;
    #pragma unroll
    for (int e = 0; e < 16; ++e) { au0[e] = f2bf(af[e]); au1[e] = f2bf(af[16 + e]); }
    const v16bf a0 = __builtin_bit_cast(v16bf, au0);
    const v16bf a1 = __builtin_bit_cast(v16bf, au1);

    // ---- running argmin of  d = ||e||^2 - 2*(z.e)  over all 512 codes ----
    float best[8];
    int   bidx[8];
    #pragma unroll
    for (int r = 0; r < 8; ++r) { best[r] = 3.4e38f; bidx[r] = 0; }

    for (int nt = 0; nt < KCODES / 16; ++nt) {
        const int n = nt * 16 + li;
        const unsigned short* bp = embLds + n * EMB_STRIDE;
        // B 32x16 layout: half h holds K = 16h..16h+15 (contiguous 32B -> 2x ds_load_b128)
        v16bf b0 = __builtin_bit_cast(v16bf, *(const v16u*)(bp + 16 * h));
        v16bf b1 = __builtin_bit_cast(v16bf, *(const v16u*)(bp + 32 + 16 * h));

        v8f c = {};
        c = __builtin_amdgcn_wmma_f32_16x16x32_bf16(false, a0, false, b0,
                                                    (short)0, c, false, false);
        c = __builtin_amdgcn_wmma_f32_16x16x32_bf16(false, a1, false, b1,
                                                    (short)0, c, false, false);
        const float en = enorm[n];
        #pragma unroll
        for (int r = 0; r < 8; ++r) {            // C layout: VGPR r -> row r + 8h
            float d = en - 2.0f * c[r];
            bool  t = d < best[r];               // branch-free select form
            best[r] = t ? d : best[r];
            bidx[r] = t ? n : bidx[r];
        }
    }

    // ---- argmin across the 16 lane-columns (stays within each half-wave) ----
    #pragma unroll
    for (int off = 1; off < 16; off <<= 1) {
        #pragma unroll
        for (int r = 0; r < 8; ++r) {
            float ov = __shfl_xor(best[r], off, 32);
            int   oi = __shfl_xor(bidx[r], off, 32);
            bool  t  = (ov < best[r]) || (ov == best[r] && oi < bidx[r]);
            best[r] = t ? ov : best[r];
            bidx[r] = t ? oi : bidx[r];
        }
    }
    // swap halves so every lane knows all 16 row winners
    int oidx[8];
    #pragma unroll
    for (int r = 0; r < 8; ++r) oidx[r] = __shfl_xor(bidx[r], 16, 32);

    // ---- gather fp32 codes, store to BOTH output halves (all b128, coalesced)
    // 16 lanes cover one 256B row with float4; the two half-waves do rows m, m+1.
    // All bidx/oidx subscripts are compile-time constants; only the final h-pick
    // is a runtime select (one v_cndmask), so arrays stay in VGPRs.
    float* out0 = out;
    float* out1 = out + (long long)nRows * DIM;
    const int c4 = li * 4;                          // column quantum within a row
    #pragma unroll
    for (int m = 0; m < 16; m += 2) {
        const int g  = m >> 3;                      // constant: group of both rows
        const int r0 = m & 7;                       // constant (even)
        const int r1 = r0 + 1;                      // constant
        int idx_h0 = (g == 0) ? bidx[r0] : oidx[r0];   // winner of row m   (h==0 lanes)
        int idx_h1 = (g == 1) ? bidx[r1] : oidx[r1];   // winner of row m+1 (h==1 lanes)
        int idx = h ? idx_h1 : idx_h0;
        int row = m + h;                            // lanes 0-15 -> row m, 16-31 -> m+1
        const float4 v = *(const float4*)(emb + idx * DIM + c4);
        long long o = (row0 + row) * DIM + c4;
        *(float4*)(out0 + o) = v;
        *(float4*)(out1 + o) = v;
    }
}

extern "C" void kernel_launch(void* const* d_in, const int* in_sizes, int n_in,
                              void* d_out, int out_size, void* d_ws, size_t ws_size,
                              hipStream_t stream) {
    (void)n_in; (void)out_size; (void)d_ws; (void)ws_size;
    const float* z   = (const float*)d_in[0];
    const float* emb = (const float*)d_in[1];
    float* out = (float*)d_out;

    const int nRows = in_sizes[0] / DIM;                 // 524288
    const int grid  = nRows / ROWS_PER_BLOCK;            // 4096
    const size_t smem = (size_t)KCODES * EMB_STRIDE * 2  // bf16 codebook (padded)
                      + (size_t)KCODES * 4;              // ||e||^2

    hipLaunchKernelGGL(vq_argmin_gather_kernel,
                       dim3(grid), dim3(BLOCK_THREADS), smem, stream,
                       z, emb, out, nRows);
}
